// GraphEncoder_76879914598858
// MI455X (gfx1250) — matrix operand order repaired and checked by reading, not compiled, not run on hardware
//
#include <hip/hip_runtime.h>
#include <hip/hip_bf16.h>
#include <stdint.h>

// ---------------------------------------------------------------------------
// MI455X (gfx1250) GNN forward: GCNConv + TransformerConv + LN + mean-pool.
// Memory-bound (edge scatter/gather ~5 GB @ 23.3 TB/s); all dense GEMMs run
// through one bf16 WMMA kernel (v_wmma_f32_16x16x32_bf16, f32 accumulate),
// 4 N-tiles per wave with pre-converted bf16 [Nc][K] weights so every B
// fragment is one aligned 32B contiguous load.
// Edge embedding e = edge_attr@We is algebraically folded away:
//   logits:  q·e   -> (q@We^T)·edge_attr + q·be
//   output:  Σα(v+e) -> Σα v + (Σα edge_attr)@We + (Σα)be
// ---------------------------------------------------------------------------

typedef __bf16 v16bf __attribute__((ext_vector_type(16)));
typedef float  v8f   __attribute__((ext_vector_type(8)));

// ---------------- utility kernels ----------------

__global__ void fill_kernel(float* __restrict__ p, float v, long n) {
  long i = (long)blockIdx.x * blockDim.x + threadIdx.x;
  if (i < n) p[i] = v;
}

__global__ void deg_kernel(const int* __restrict__ ei, float* __restrict__ deg, int E) {
  int e = blockIdx.x * blockDim.x + threadIdx.x;
  if (e < E) atomicAdd(&deg[ei[E + e]], 1.0f);
}

__global__ void dinv_kernel(float* __restrict__ dinv, int N) {
  int i = blockIdx.x * blockDim.x + threadIdx.x;
  if (i < N) dinv[i] = rsqrtf(1.0f + dinv[i]);
}

__global__ void relu_kernel(float* __restrict__ p, long n) {
  long i = (long)blockIdx.x * blockDim.x + threadIdx.x;
  if (i < n) p[i] = fmaxf(p[i], 0.0f);
}

// Convert weight to bf16 in [Nc][K] layout.
// trans=0: source W is [K][Nc] row-major -> Wt[n*K+k] = W[k*Nc+n]
// trans=1: source W is already [Nc][K]   -> Wt[i]     = W[i]
__global__ void conv_w_kernel(const float* __restrict__ W, __bf16* __restrict__ Wt,
                              int K, int Nc, int trans) {
  int i = blockIdx.x * blockDim.x + threadIdx.x;
  if (i >= K * Nc) return;
  if (trans) {
    Wt[i] = (__bf16)W[i];
  } else {
    int n = i / K, kk = i - n * K;
    Wt[i] = (__bf16)W[kk * Nc + n];
  }
}

// ---------------- bf16 WMMA GEMM ----------------
// C[M,Nc] = A[M,K] @ W + epilogue, with Wt = bf16 W^T stored [Nc][K].
// One wave computes a 16x64 output strip: one A fragment per K-step feeds
// 4 WMMAs into 4 independent accumulators. flags: 2 = accumulate into C.
// epilogue bias term: rowScale ? rowScale[row]*bias[col] : bias[col].
__global__ void gemm_bf16_wmma(const float* __restrict__ A, const __bf16* __restrict__ Wt,
                               const float* __restrict__ bias, const float* __restrict__ rowScale,
                               float* __restrict__ C, int M, int K, int Nc, int flags) {
  const int groupsN = Nc >> 6;                       // 4 tiles of 16 cols per group
  const int tilesM  = (M + 15) >> 4;
  int wid = (blockIdx.x * blockDim.x + threadIdx.x) >> 5;
  if (wid >= tilesM * groupsN) return;               // whole-wave exit: EXEC stays all-ones
  const int tM = wid / groupsN;
  const int gN = wid - tM * groupsN;
  const int lane = threadIdx.x & 31;
  const int l15  = lane & 15;
  const int hi   = lane >> 4;

  int arowIdx = tM * 16 + l15; if (arowIdx >= M) arowIdx = M - 1;  // clamp, no divergence
  const float* __restrict__ arow = A + (long)arowIdx * K;

  v8f acc[4];
#pragma unroll
  for (int t = 0; t < 4; ++t) acc[t] = (v8f){0.f,0.f,0.f,0.f,0.f,0.f,0.f,0.f};

  for (int kk = 0; kk < K; kk += 32) {
    // A frag: lanes 0-15 = M rows; elems i -> K = hi*16 + i (matches B order)
    v16bf a;
#pragma unroll
    for (int i = 0; i < 8; ++i) {
      a[i]     = (__bf16)arow[kk + hi * 16 + i];
      a[8 + i] = (__bf16)arow[kk + hi * 16 + 8 + i];
    }
#pragma unroll
    for (int t = 0; t < 4; ++t) {
      // B frag: lane = N-col; elems i -> K = hi*16 + i  (one aligned 32B load)
      const int col = gN * 64 + t * 16 + l15;
      v16bf b = *(const v16bf*)(Wt + (long)col * K + kk + hi * 16);
      acc[t] = __builtin_amdgcn_wmma_f32_16x16x32_bf16(false, a, false, b, (short)0,
                                                       acc[t], false, false);
    }
  }

  // D frag: lane L, elem r -> row = (L>>4)*8 + r, col = L&15
#pragma unroll
  for (int t = 0; t < 4; ++t) {
    const int col = gN * 64 + t * 16 + l15;
    const float bv = bias ? bias[col] : 0.0f;
#pragma unroll
    for (int r = 0; r < 8; ++r) {
      int orow = tM * 16 + hi * 8 + r;
      if (orow >= M) continue;
      long off = (long)orow * Nc + col;
      float val = acc[t][r] + (rowScale ? rowScale[orow] * bv : bv);
      if (flags & 2) val += C[off];
      C[off] = val;
    }
  }
}

// ---------------- GCN aggregation ----------------

// agg[n][c] = xw[n][c]*dinv[n]^2 + b1[c]   (self-loop term + bias, pre-scatter)
__global__ void gcn_init_kernel(const float* __restrict__ xw, const float* __restrict__ dinv,
                                const float* __restrict__ b1, float* __restrict__ agg, int N) {
  long i = (long)blockIdx.x * blockDim.x + threadIdx.x;
  if (i >= (long)N * 256) return;
  int row = (int)(i >> 8), colc = (int)(i & 255);
  float di = dinv[row];
  agg[i] = xw[i] * di * di + b1[colc];
}

// one wave per edge: agg[dst] += xw[src] * dinv[src]*dinv[dst]   (256 ch, 8/lane)
__global__ void gcn_scatter_kernel(const int* __restrict__ ei, const float* __restrict__ xw,
                                   const float* __restrict__ dinv, float* __restrict__ agg,
                                   int E) {
  int wid = (blockIdx.x * blockDim.x + threadIdx.x) >> 5;
  if (wid >= E) return;
  int lane = threadIdx.x & 31;
  int s = ei[wid], d = ei[E + wid];
  float nrm = dinv[s] * dinv[d];
  const float* __restrict__ xr = xw + (long)s * 256;
  float* __restrict__ ar = agg + (long)d * 256;
#pragma unroll
  for (int j = 0; j < 8; ++j) {
    int c = lane + 32 * j;
    atomicAdd(&ar[c], xr[c] * nrm);
  }
}

// ---------------- attention ----------------

__global__ void qbe_kernel(const float* __restrict__ q, const float* __restrict__ be,
                           float* __restrict__ qbe, int N) {
  int n = blockIdx.x * blockDim.x + threadIdx.x;
  if (n >= N) return;
  const float* __restrict__ qr = q + (long)n * 128;
  float s = 0.0f;
  for (int c = 0; c < 128; ++c) s += qr[c] * be[c];
  qbe[n] = s;
}

// one wave per edge: logits = (q[dst]·k[src] + qe[dst]·ea[e] + qbe[dst]) / sqrt(128)
__global__ void logits_kernel(const int* __restrict__ ei, const float* __restrict__ q,
                              const float* __restrict__ k, const float* __restrict__ qe,
                              const float* __restrict__ ea, const float* __restrict__ qbe,
                              float* __restrict__ logits, float* __restrict__ lmax, int E) {
  int wid = (blockIdx.x * blockDim.x + threadIdx.x) >> 5;
  if (wid >= E) return;
  int lane = threadIdx.x & 31;
  int s = ei[wid], d = ei[E + wid];
  const float* __restrict__ qr  = q  + (long)d * 128;
  const float* __restrict__ kr  = k  + (long)s * 128;
  const float* __restrict__ qer = qe + (long)d * 64;
  const float* __restrict__ ear = ea + (long)wid * 64;
  float p = 0.0f;
#pragma unroll
  for (int j = 0; j < 4; ++j) { int c = lane + 32 * j; p += qr[c] * kr[c]; }
#pragma unroll
  for (int j = 0; j < 2; ++j) { int u = lane + 32 * j; p += qer[u] * ear[u]; }
#pragma unroll
  for (int off = 16; off > 0; off >>= 1) p += __shfl_xor(p, off, 32);
  if (lane == 0) {
    float l = (p + qbe[d]) * 0.08838834764831845f;   // 1/sqrt(128)
    logits[wid] = l;
    __hip_atomic_fetch_max(&lmax[d], l, __ATOMIC_RELAXED, __HIP_MEMORY_SCOPE_AGENT);
  }
}

__global__ void softmax_exp_kernel(const int* __restrict__ ei, const float* __restrict__ logits,
                                   const float* __restrict__ lmax, float* __restrict__ expl,
                                   float* __restrict__ denom, int E) {
  int e = blockIdx.x * blockDim.x + threadIdx.x;
  if (e >= E) return;
  int d = ei[E + e];
  float x = __expf(logits[e] - lmax[d]);
  expl[e] = x;
  atomicAdd(&denom[d], x);
}

// one wave per edge: outb[dst]+=a*v[src]; svec[dst]+=a*ea[e]; salpha[dst]+=a
__global__ void attn_scatter_kernel(const int* __restrict__ ei, const float* __restrict__ expl,
                                    const float* __restrict__ denom, const float* __restrict__ v,
                                    const float* __restrict__ ea, float* __restrict__ outb,
                                    float* __restrict__ svec, float* __restrict__ salpha, int E) {
  int wid = (blockIdx.x * blockDim.x + threadIdx.x) >> 5;
  if (wid >= E) return;
  int lane = threadIdx.x & 31;
  int s = ei[wid], d = ei[E + wid];
  float alpha = expl[wid] / denom[d];      // denom>0: this edge contributed
  const float* __restrict__ vr = v + (long)s * 128;
  float* __restrict__ orow = outb + (long)d * 128;
#pragma unroll
  for (int j = 0; j < 4; ++j) { int c = lane + 32 * j; atomicAdd(&orow[c], alpha * vr[c]); }
  const float* __restrict__ ear = ea + (long)wid * 64;
  float* __restrict__ sr = svec + (long)d * 64;
#pragma unroll
  for (int j = 0; j < 2; ++j) { int u = lane + 32 * j; atomicAdd(&sr[u], alpha * ear[u]); }
  if (lane == 0) atomicAdd(&salpha[d], alpha);
}

// ---------------- LayerNorm + ReLU + mean-pool ----------------

__global__ void ln_pool_kernel(const float* __restrict__ outb, const int* __restrict__ batch,
                               const float* __restrict__ gamma, const float* __restrict__ beta,
                               float* __restrict__ psum, float* __restrict__ pcnt, int N) {
  int wid = (blockIdx.x * blockDim.x + threadIdx.x) >> 5;
  if (wid >= N) return;
  int lane = threadIdx.x & 31;
  const float* __restrict__ orow = outb + (long)wid * 128;
  float vals[4];
  float s = 0.0f;
#pragma unroll
  for (int j = 0; j < 4; ++j) { vals[j] = orow[lane + 32 * j]; s += vals[j]; }
#pragma unroll
  for (int off = 16; off > 0; off >>= 1) s += __shfl_xor(s, off, 32);
  float mu = s * (1.0f / 128.0f);
  float vsum = 0.0f;
#pragma unroll
  for (int j = 0; j < 4; ++j) { float dd = vals[j] - mu; vsum += dd * dd; }
#pragma unroll
  for (int off = 16; off > 0; off >>= 1) vsum += __shfl_xor(vsum, off, 32);
  float inv = rsqrtf(vsum * (1.0f / 128.0f) + 1e-5f);
  int b = batch[wid];
  float* __restrict__ ps = psum + (long)b * 128;
#pragma unroll
  for (int j = 0; j < 4; ++j) {
    int c = lane + 32 * j;
    float o = (vals[j] - mu) * inv * gamma[c] + beta[c];
    o = fmaxf(o, 0.0f);
    atomicAdd(&ps[c], o);
  }
  if (lane == 0) atomicAdd(&pcnt[b], 1.0f);
}

__global__ void pool_div_kernel(const float* __restrict__ psum, const float* __restrict__ pcnt,
                                float* __restrict__ out, int B) {
  int i = blockIdx.x * blockDim.x + threadIdx.x;
  if (i >= B * 128) return;
  out[i] = psum[i] / fmaxf(pcnt[i >> 7], 1.0f);
}

// ---------------- host launcher ----------------

extern "C" void kernel_launch(void* const* d_in, const int* in_sizes, int n_in,
                              void* d_out, int out_size, void* d_ws, size_t ws_size,
                              hipStream_t stream) {
  (void)n_in; (void)ws_size;
  const float* x     = (const float*)d_in[0];
  const float* ea    = (const float*)d_in[1];
  const int*   ei    = (const int*)  d_in[2];
  const int*   batch = (const int*)  d_in[3];
  const float* W1 = (const float*)d_in[4];  const float* b1 = (const float*)d_in[5];
  const float* Wq = (const float*)d_in[6];  const float* bq = (const float*)d_in[7];
  const float* Wk = (const float*)d_in[8];  const float* bk = (const float*)d_in[9];
  const float* Wv = (const float*)d_in[10]; const float* bv = (const float*)d_in[11];
  const float* We = (const float*)d_in[12]; const float* be = (const float*)d_in[13];
  const float* Wr = (const float*)d_in[14]; const float* br = (const float*)d_in[15];
  const float* gamma = (const float*)d_in[16];
  const float* beta  = (const float*)d_in[17];
  float* out = (float*)d_out;

  const int N = in_sizes[0] / 128;
  const int E = in_sizes[1] / 64;
  const int B = out_size / 128;

  // workspace layout (floats)
  float* p = (float*)d_ws;
  float* xw     = p; p += (long)N * 256;
  float* h      = p; p += (long)N * 256;   // agg, then h (in-place relu)
  float* dinv   = p; p += N;               // deg, then rsqrt in-place
  float* qbuf   = p; p += (long)N * 128;
  float* kbuf   = p; p += (long)N * 128;
  float* vbuf   = p; p += (long)N * 128;
  float* qe     = p; p += (long)N * 64;
  float* qbe    = p; p += N;
  float* outb   = p; p += (long)N * 128;
  float* svec   = p; p += (long)N * 64;
  float* salpha = p; p += N;
  float* logit  = p; p += E;
  float* expl   = p; p += E;
  float* lmax   = p; p += N;
  float* denom  = p; p += N;
  float* psum   = p; p += (long)B * 128;
  float* pcnt   = p; p += B;               // contiguous after psum

  // bf16 weight area (64B aligned for v16bf loads)
  uintptr_t wbase = ((uintptr_t)p + 63) & ~(uintptr_t)63;
  __bf16* wb  = (__bf16*)wbase;
  __bf16* W1t = wb; wb += 256 * 128;       // [Nc=256][K=128]
  __bf16* Wqt = wb; wb += 128 * 256;       // [Nc=128][K=256]
  __bf16* Wkt = wb; wb += 128 * 256;
  __bf16* Wvt = wb; wb += 128 * 256;
  __bf16* Wrt = wb; wb += 128 * 256;
  __bf16* WeQ = wb; wb += 64 * 128;        // qe GEMM: [Nc=64][K=128] == We layout
  __bf16* WeS = wb; wb += 128 * 64;        // svec GEMM: [Nc=128][K=64] == We^T

  auto cdiv = [](long a, long b) { return (int)((a + b - 1) / b); };
  auto gemm = [&](const float* A, const __bf16* Wt, const float* bias, const float* rs,
                  float* C, int M, int K, int Nc, int flags) {
    int waves = ((M + 15) >> 4) * (Nc >> 6);
    gemm_bf16_wmma<<<cdiv(waves, 8), 256, 0, stream>>>(A, Wt, bias, rs, C, M, K, Nc, flags);
  };

  // ---- weight prep (tiny) ----
  conv_w_kernel<<<cdiv(128 * 256, 256), 256, 0, stream>>>(W1, W1t, 128, 256, 0);
  conv_w_kernel<<<cdiv(256 * 128, 256), 256, 0, stream>>>(Wq, Wqt, 256, 128, 0);
  conv_w_kernel<<<cdiv(256 * 128, 256), 256, 0, stream>>>(Wk, Wkt, 256, 128, 0);
  conv_w_kernel<<<cdiv(256 * 128, 256), 256, 0, stream>>>(Wv, Wvt, 256, 128, 0);
  conv_w_kernel<<<cdiv(256 * 128, 256), 256, 0, stream>>>(Wr, Wrt, 256, 128, 0);
  conv_w_kernel<<<cdiv(64 * 128, 256), 256, 0, stream>>>(We, WeQ, 128, 64, 1);
  conv_w_kernel<<<cdiv(64 * 128, 256), 256, 0, stream>>>(We, WeS, 64, 128, 0);

  // ---- init scratch ----
  fill_kernel<<<cdiv(N, 256), 256, 0, stream>>>(dinv, 0.0f, N);
  fill_kernel<<<cdiv(N, 256), 256, 0, stream>>>(lmax, -3.0e38f, N);
  fill_kernel<<<cdiv(N, 256), 256, 0, stream>>>(denom, 0.0f, N);
  fill_kernel<<<cdiv((long)N * 64, 256), 256, 0, stream>>>(svec, 0.0f, (long)N * 64);
  fill_kernel<<<cdiv(N, 256), 256, 0, stream>>>(salpha, 0.0f, N);
  fill_kernel<<<cdiv((long)B * 128 + B, 256), 256, 0, stream>>>(psum, 0.0f, (long)B * 128 + B);

  // ---- degrees ----
  deg_kernel<<<cdiv(E, 256), 256, 0, stream>>>(ei, dinv, E);
  dinv_kernel<<<cdiv(N, 256), 256, 0, stream>>>(dinv, N);

  // ---- GCNConv ----
  gemm(x, W1t, nullptr, nullptr, xw, N, 128, 256, 0);                      // xw = x@W1
  gcn_init_kernel<<<cdiv((long)N * 256, 256), 256, 0, stream>>>(xw, dinv, b1, h, N);
  gcn_scatter_kernel<<<cdiv((long)E * 32, 256), 256, 0, stream>>>(ei, xw, dinv, h, E);
  relu_kernel<<<cdiv((long)N * 256, 256), 256, 0, stream>>>(h, (long)N * 256);

  // ---- TransformerConv projections ----
  gemm(h, Wqt, bq, nullptr, qbuf, N, 256, 128, 0);
  gemm(h, Wkt, bk, nullptr, kbuf, N, 256, 128, 0);
  gemm(h, Wvt, bv, nullptr, vbuf, N, 256, 128, 0);
  gemm(h, Wrt, br, nullptr, outb, N, 256, 128, 0);                         // root/skip
  gemm(qbuf, WeQ, nullptr, nullptr, qe, N, 128, 64, 0);                    // qe = q@We^T
  qbe_kernel<<<cdiv(N, 256), 256, 0, stream>>>(qbuf, be, qbe, N);

  // ---- edge attention ----
  logits_kernel<<<cdiv((long)E * 32, 256), 256, 0, stream>>>(ei, qbuf, kbuf, qe, ea, qbe,
                                                             logit, lmax, E);
  softmax_exp_kernel<<<cdiv(E, 256), 256, 0, stream>>>(ei, logit, lmax, expl, denom, E);
  attn_scatter_kernel<<<cdiv((long)E * 32, 256), 256, 0, stream>>>(ei, expl, denom, vbuf, ea,
                                                                   outb, svec, salpha, E);
  gemm(svec, WeS, be, salpha, outb, N, 64, 128, 2);      // outb += svec@We + salpha*be

  // ---- LayerNorm + ReLU + mean pool ----
  ln_pool_kernel<<<cdiv((long)N * 32, 256), 256, 0, stream>>>(outb, batch, gamma, beta,
                                                              psum, pcnt, N);
  pool_div_kernel<<<cdiv((long)B * 128, 256), 256, 0, stream>>>(psum, pcnt, out, B);
}
